// UnifiedEmbedding_77360950936122
// MI455X (gfx1250) — matrix operand order repaired and checked by reading, not compiled, not run on hardware
//
#include <hip/hip_runtime.h>
#include <hip/hip_bf16.h>

typedef __attribute__((ext_vector_type(16))) __bf16 v16bf;
typedef __attribute__((ext_vector_type(8)))  float  v8f;

#define FRONT 3072
#define EMBED 768
#define NTOK  (4 * 4096)
#define TM    64            // tokens per workgroup (4 m-tiles)
#define MT    4             // m-tiles per wave
#define NT    3             // n-tiles per wave  (16 waves * 3 = 48)
#define KC    256           // K-chunk staged in LDS for stage 1
#define NWAVE 16            // 512 threads

static __device__ __forceinline__ unsigned short f32_to_bf16(float f) {
    // round-to-nearest-even truncation
    unsigned int u = __builtin_bit_cast(unsigned int, f);
    unsigned int r = u + 0x7FFFu + ((u >> 16) & 1u);
    return (unsigned short)(r >> 16);
}

union Frag { v16bf v; uint4 q[2]; };

// ---------------------------------------------------------------------------
// One-shot f32 -> bf16 weight conversion into workspace (vectorized x4).
// ---------------------------------------------------------------------------
__global__ __launch_bounds__(256) void cvt_f32_to_bf16(
    const float* __restrict__ src, unsigned short* __restrict__ dst, int n4) {
    int i = blockIdx.x * 256 + threadIdx.x;
    if (i < n4) {
        float4 v = ((const float4*)src)[i];
        ushort4 d;
        d.x = f32_to_bf16(v.x);
        d.y = f32_to_bf16(v.y);
        d.z = f32_to_bf16(v.z);
        d.w = f32_to_bf16(v.w);
        ((ushort4*)dst)[i] = d;
    }
}

// ---------------------------------------------------------------------------
// Fused: gather -> GEMM1(bf16 wmma, f32 acc) -> bias -> exact GeLU ->
//        GEMM2(bf16 wmma, f32 acc) -> bias -> f32 store
// Workgroup: 512 threads (16 waves), 64 tokens.
// Each wave: 4 m-tiles x 3 n-tiles -> each B fragment feeds 4 WMMAs.
// ---------------------------------------------------------------------------
__global__ __launch_bounds__(512) void fused_embed_mlp(
    const int*            __restrict__ idxs,
    const float*          __restrict__ tok_embed,
    const float*          __restrict__ b1,
    const float*          __restrict__ b2,
    const unsigned short* __restrict__ W1b,   // [EMBED][FRONT] bf16
    const unsigned short* __restrict__ W2b,   // [EMBED][EMBED] bf16
    float*                __restrict__ out)   // [NTOK][EMBED]
{
    __shared__ __align__(16) unsigned short s_a[TM * KC];     // act chunk (bf16), 32 KB
    __shared__ __align__(16) unsigned short s_h[TM * EMBED];  // hidden (bf16), 96 KB
    __shared__ int s_tok[TM];

    const int tid  = threadIdx.x;
    const int lane = tid & 31;
    const int wave = tid >> 5;
    const int half = lane >> 4;   // 0 for lanes 0-15, 1 for lanes 16-31
    const int l16  = lane & 15;

    const int tok_base = blockIdx.x * TM;

    if (tid < TM) s_tok[tid] = idxs[tok_base + tid];

    v8f acc[MT * NT];                      // [mt][t] -> acc[mt*NT+t]
    #pragma unroll
    for (int j = 0; j < MT * NT; ++j)
        #pragma unroll
        for (int i = 0; i < 8; ++i) acc[j][i] = 0.0f;

    // ---------------- Stage 1: hidden = gather(x) @ W1^T ----------------
    #pragma unroll 1
    for (int kc = 0; kc < FRONT / KC; ++kc) {
        __syncthreads();  // also covers s_tok on first iteration
        // cooperative gather + f32->bf16 convert: TM x KC elements, float4 at a time
        for (int q = tid; q < TM * KC / 4; q += NWAVE * 32) {
            const int row  = q >> 6;            // q / (KC/4), KC/4 == 64
            const int col4 = (q & 63) << 2;
            const float4 v = *(const float4*)(tok_embed +
                                (size_t)s_tok[row] * FRONT + kc * KC + col4);
            ushort4 d;
            d.x = f32_to_bf16(v.x);
            d.y = f32_to_bf16(v.y);
            d.z = f32_to_bf16(v.z);
            d.w = f32_to_bf16(v.w);
            *(ushort4*)&s_a[row * KC + col4] = d;
        }
        __syncthreads();

        #pragma unroll 1
        for (int k0 = 0; k0 < KC; k0 += 32) {
            // A fragments: lane = row l16 of each m-tile, K = k0 + 8*half + {0..7}, +16
            Frag a[MT];
            #pragma unroll
            for (int mt = 0; mt < MT; ++mt) {
                const unsigned short* ap = &s_a[(mt * 16 + l16) * KC + k0 + half * 8];
                a[mt].q[0] = *(const uint4*)(ap);
                a[mt].q[1] = *(const uint4*)(ap + 16);
            }
            #pragma unroll
            for (int t = 0; t < NT; ++t) {
                const int nt = wave * NT + t;
                // B fragment: lane = col l16, K = k0 + 8*half + {0..7}, +16
                Frag b;
                const unsigned short* bp = W1b + (size_t)(nt * 16 + l16) * FRONT
                                               + kc * KC + k0 + half * 8;
                b.q[0] = *(const uint4*)(bp);
                b.q[1] = *(const uint4*)(bp + 16);
                #pragma unroll
                for (int mt = 0; mt < MT; ++mt)
                    acc[mt * NT + t] = __builtin_amdgcn_wmma_f32_16x16x32_bf16(
                        false, a[mt].v, false, b.v, (short)0, acc[mt * NT + t],
                        false, false);
            }
        }
    }

    // bias + exact GeLU, write hidden to LDS as bf16
    #pragma unroll
    for (int t = 0; t < NT; ++t) {
        const int   n    = (wave * NT + t) * 16 + l16;
        const float bias = b1[n];
        #pragma unroll
        for (int mt = 0; mt < MT; ++mt) {
            #pragma unroll
            for (int v = 0; v < 8; ++v) {
                const int   m = mt * 16 + v + half * 8;
                const float x = acc[mt * NT + t][v] + bias;
                const float g = 0.5f * x * (1.0f + erff(x * 0.70710678118654752f));
                s_h[m * EMBED + n] = f32_to_bf16(g);
            }
        }
    }
    __syncthreads();

    // ---------------- Stage 2: out = gelu(hidden) @ W2^T + b2 ----------------
    v8f acc2[MT * NT];
    #pragma unroll
    for (int j = 0; j < MT * NT; ++j)
        #pragma unroll
        for (int i = 0; i < 8; ++i) acc2[j][i] = 0.0f;

    #pragma unroll 1
    for (int k0 = 0; k0 < EMBED; k0 += 32) {
        Frag a[MT];
        #pragma unroll
        for (int mt = 0; mt < MT; ++mt) {
            const unsigned short* ap = &s_h[(mt * 16 + l16) * EMBED + k0 + half * 8];
            a[mt].q[0] = *(const uint4*)(ap);
            a[mt].q[1] = *(const uint4*)(ap + 16);
        }
        #pragma unroll
        for (int t = 0; t < NT; ++t) {
            const int nt = wave * NT + t;
            Frag b;
            const unsigned short* bp = W2b + (size_t)(nt * 16 + l16) * EMBED
                                           + k0 + half * 8;
            b.q[0] = *(const uint4*)(bp);
            b.q[1] = *(const uint4*)(bp + 16);
            #pragma unroll
            for (int mt = 0; mt < MT; ++mt)
                acc2[mt * NT + t] = __builtin_amdgcn_wmma_f32_16x16x32_bf16(
                    false, a[mt].v, false, b.v, (short)0, acc2[mt * NT + t],
                    false, false);
        }
    }

    // epilogue: bias + f32 store
    #pragma unroll
    for (int t = 0; t < NT; ++t) {
        const int   n    = (wave * NT + t) * 16 + l16;
        const float bias = b2[n];
        #pragma unroll
        for (int mt = 0; mt < MT; ++mt) {
            #pragma unroll
            for (int v = 0; v < 8; ++v) {
                const int m = mt * 16 + v + half * 8;
                out[(size_t)(tok_base + m) * EMBED + n] = acc2[mt * NT + t][v] + bias;
            }
        }
    }
}

extern "C" void kernel_launch(void* const* d_in, const int* in_sizes, int n_in,
                              void* d_out, int out_size, void* d_ws, size_t ws_size,
                              hipStream_t stream) {
    const int*   idxs      = (const int*)d_in[0];
    const float* tok_embed = (const float*)d_in[1];
    const float* W1        = (const float*)d_in[2];
    const float* b1        = (const float*)d_in[3];
    const float* W2        = (const float*)d_in[4];
    const float* b2        = (const float*)d_in[5];
    float*       out       = (float*)d_out;

    unsigned short* W1b = (unsigned short*)d_ws;                 // 768*3072 bf16
    unsigned short* W2b = W1b + (size_t)EMBED * FRONT;           // 768*768  bf16

    const int n4_w1 = EMBED * FRONT / 4;   // 589824
    const int n4_w2 = EMBED * EMBED / 4;   // 147456
    cvt_f32_to_bf16<<<(n4_w1 + 255) / 256, 256, 0, stream>>>(W1, W1b, n4_w1);
    cvt_f32_to_bf16<<<(n4_w2 + 255) / 256, 256, 0, stream>>>(W2, W2b, n4_w2);

    fused_embed_mlp<<<NTOK / TM, 512, 0, stream>>>(
        idxs, tok_embed, b1, b2, W1b, W2b, out);
}
